// GumbelSoftmaxTokenizer_50165218017900
// MI455X (gfx1250) — matrix operand order repaired and checked by reading, not compiled, not run on hardware
//
#include <hip/hip_runtime.h>
#include <hip/hip_bf16.h>
#include <math.h>

// ---------------------------------------------------------------------------
// GumbelSoftmaxTokenizer for MI455X (gfx1250, wave32, WMMA).
#define BATCH   16
#define PPTS    8192
#define NPTS    (BATCH * PPTS)      // 131072
#define FEAT    32
#define TOKD    768
#define MAXTOK  128
#define KNN_K   16
#define IMPH    256
// All dense layers run as bf16 WMMA GEMMs (v_wmma_f32_16x16x32_bf16) with
// f32 accumulation; activations kept in bf16 to halve HBM traffic.
// GEMM: 256x64 block tile, 32x64 wave tile (8 WMMA/K-step/wave),
// double-buffered LDS; A tile staged with GLOBAL_LOAD_ASYNC_TO_LDS_B128
// (ASYNCcnt-tracked DMA) when the builtin is available.
// ---------------------------------------------------------------------------

typedef __attribute__((ext_vector_type(16))) __bf16 v16bf;
typedef __attribute__((ext_vector_type(8)))  float  v8f;
typedef __attribute__((ext_vector_type(4)))  int    v4i;

union AF { v16bf v; uint4 q[2]; };

#if defined(__AMDGCN__) && \
    __has_builtin(__builtin_amdgcn_global_load_async_to_lds_b128) && \
    __has_builtin(__builtin_amdgcn_s_wait_asynccnt)
#define HAVE_ASYNC_LDS 1
// Builtin signature (from hipcc diagnostic): (v4i as(1)*, v4i as(3)*, Ii, Ii)
#define GAS128(p) ((__attribute__((address_space(1))) v4i*)(p))
#define LAS128(p) ((__attribute__((address_space(3))) v4i*)(p))
#else
#define HAVE_ASYNC_LDS 0
#endif

static __device__ __forceinline__ unsigned short f32_to_bf16(float f) {
  unsigned u = __float_as_uint(f);
  unsigned r = u + 0x7FFFu + ((u >> 16) & 1u);   // round-to-nearest-even
  return (unsigned short)(r >> 16);
}
static __device__ __forceinline__ float bf16_to_f32(unsigned short h) {
  return __uint_as_float(((unsigned)h) << 16);
}

// --------------------------- f32 -> bf16 convert ---------------------------
__global__ void cvt_f32_bf16(const float* __restrict__ in,
                             unsigned short* __restrict__ out, long n) {
  long i = (long)blockIdx.x * blockDim.x + threadIdx.x;
  long stride = (long)gridDim.x * blockDim.x;
  for (; i < n; i += stride) out[i] = f32_to_bf16(in[i]);
}

// ------------------------------- WMMA GEMM ---------------------------------
// C[M,N] = act( A[M,K](bf16) @ W[K,N](bf16) + bias + coord-tail )
#define TM 256   // block rows: 8 waves x 32 rows
#define TN 64    // block cols: 4 wmma col-tiles
#define TKS 32
#define LDP 40   // padded LDS row (ushorts): 80B = 20 banks, 16B-aligned

__global__ __launch_bounds__(256) void gemm_bf16_wmma(
    const unsigned short* __restrict__ A,   // M x Kd, row-major bf16
    const unsigned short* __restrict__ W,   // Kd x Nd, row-major bf16
    const float* __restrict__ bias,         // Nd (may be null)
    const float* __restrict__ coord5,       // NPTS x 5 f32 (may be null)
    const float* __restrict__ wtail,        // 4 x Nd f32 (may be null)
    unsigned short* __restrict__ obf,       // bf16 out (may be null)
    float* __restrict__ of32,               // f32 out (may be null)
    int Kd, int Nd, int relu) {
  __shared__ __align__(16) unsigned short sA[2][TM][LDP];
  __shared__ __align__(16) unsigned short sB[2][TN][LDP];

  const int t    = threadIdx.x;
  const int wv   = t >> 5;
  const int lane = t & 31;
  const int hh   = lane >> 4;     // wave half
  const int lm   = lane & 15;
  const int cb   = blockIdx.x * TN;
  const long rb  = (long)blockIdx.y * TM;

  v8f acc[2][4];
#pragma unroll
  for (int rg = 0; rg < 2; ++rg)
#pragma unroll
    for (int ct = 0; ct < 4; ++ct)
#pragma unroll
      for (int i = 0; i < 8; ++i) acc[rg][ct][i] = 0.0f;

  const int bk = t >> 3, bn0 = (t & 7) * 8;   // B staging: 8 ushorts/thread

  // Stage one 256x32 A tile + transposed 32x64 B tile into LDS buffer `buf`.
  auto stage = [&](int buf, int kb) {
#if HAVE_ASYNC_LDS
    // Async DMA straight into LDS (tracked by ASYNCcnt, no VGPR round-trip).
    const unsigned short* ga = A + (rb + t) * (long)Kd + kb;
#pragma unroll
    for (int c = 0; c < 4; ++c)
      __builtin_amdgcn_global_load_async_to_lds_b128(
          GAS128(ga + c * 8), LAS128(&sA[buf][t][c * 8]), 0, 0);
#else
    const uint4* asrc = (const uint4*)(A + (rb + t) * (long)Kd + kb);
    uint4 r0 = asrc[0], r1 = asrc[1], r2 = asrc[2], r3 = asrc[3];
    *(uint4*)&sA[buf][t][0]  = r0;
    *(uint4*)&sA[buf][t][8]  = r1;
    *(uint4*)&sA[buf][t][16] = r2;
    *(uint4*)&sA[buf][t][24] = r3;
    if (kb + TKS < Kd)  // hint next tile (global_prefetch_b8)
      __builtin_prefetch(A + (rb + t) * (long)Kd + kb + TKS, 0, 3);
#endif
    union { uint4 q; unsigned short us[8]; } bq;
    bq.q = *(const uint4*)(W + (long)(kb + bk) * Nd + cb + bn0);
#pragma unroll
    for (int j = 0; j < 8; ++j) sB[buf][bn0 + j][bk] = bq.us[j];
  };

  stage(0, 0);
#if HAVE_ASYNC_LDS
  __builtin_amdgcn_s_wait_asynccnt(0);
#endif
  __syncthreads();

  int buf = 0;
  for (int kb = 0; kb < Kd; kb += TKS) {
    const int nkb = kb + TKS;
    if (nkb < Kd) stage(buf ^ 1, nkb);   // overlap next-tile staging

    // A fragments (two 16-row groups per wave):
    // lane(0-15) row=lm K=0..7,16..23 ; lane(16-31) K=8..15,24..31
    AF fa[2];
#pragma unroll
    for (int rg = 0; rg < 2; ++rg) {
      fa[rg].q[0] = *(const uint4*)&sA[buf][wv * 32 + rg * 16 + lm][hh * 8];
      fa[rg].q[1] = *(const uint4*)&sA[buf][wv * 32 + rg * 16 + lm][hh * 8 + 16];
    }
#pragma unroll
    for (int ct = 0; ct < 4; ++ct) {
      // B fragment: lane(0-15) col=lm K=0..7,16..23 ; lane(16-31) K=8..15,24..31
      AF fb;
      fb.q[0] = *(const uint4*)&sB[buf][ct * 16 + lm][hh * 8];
      fb.q[1] = *(const uint4*)&sB[buf][ct * 16 + lm][hh * 8 + 16];
#pragma unroll
      for (int rg = 0; rg < 2; ++rg)
        acc[rg][ct] = __builtin_amdgcn_wmma_f32_16x16x32_bf16(
            false, fa[rg].v, false, fb.v, (short)0, acc[rg][ct], false, false);
    }
#if HAVE_ASYNC_LDS
    if (nkb < Kd) __builtin_amdgcn_s_wait_asynccnt(0);
#endif
    __syncthreads();
    buf ^= 1;
  }

  // Epilogue: bias + optional 4-dim coordinate tail + ReLU, bf16/f32 stores.
#pragma unroll
  for (int rg = 0; rg < 2; ++rg)
#pragma unroll
    for (int ct = 0; ct < 4; ++ct) {
      const int col = cb + ct * 16 + lm;
      const float bs = bias ? bias[col] : 0.0f;
#pragma unroll
      for (int i = 0; i < 8; ++i) {
        const long row = rb + wv * 32 + rg * 16 + hh * 8 + i;
        float v = acc[rg][ct][i] + bs;
        if (wtail) {
          const float* c4 = coord5 + row * 5 + 1;
          v += c4[0] * wtail[0 * Nd + col] + c4[1] * wtail[1 * Nd + col] +
               c4[2] * wtail[2 * Nd + col] + c4[3] * wtail[3 * Nd + col];
        }
        if (relu) v = fmaxf(v, 0.0f);
        if (obf)  obf[row * (long)Nd + col] = f32_to_bf16(v);
        if (of32) of32[row * (long)Nd + col] = v;
      }
    }
}

// ------------------------- LayerNorm over 256 feats ------------------------
__global__ __launch_bounds__(256) void ln256(const float* __restrict__ in,
                                             const float* __restrict__ gam,
                                             const float* __restrict__ bet,
                                             unsigned short* __restrict__ out) {
  __shared__ float red[256];
  __shared__ float smu, svar;
  const long r = blockIdx.x;
  const int t = threadIdx.x;
  const float x = in[r * IMPH + t];
  red[t] = x;
  __syncthreads();
  for (int off = 128; off > 0; off >>= 1) {
    if (t < off) red[t] += red[t + off];
    __syncthreads();
  }
  if (t == 0) smu = red[0] * (1.0f / IMPH);
  __syncthreads();
  const float d = x - smu;
  red[t] = d * d;
  __syncthreads();
  for (int off = 128; off > 0; off >>= 1) {
    if (t < off) red[t] += red[t + off];
    __syncthreads();
  }
  if (t == 0) svar = red[0] * (1.0f / IMPH);
  __syncthreads();
  const float y = d * rsqrtf(svar + 1e-5f) * gam[t] + bet[t];
  out[r * IMPH + t] = f32_to_bf16(y);
}

// ------------------------- importance head (dot w3) ------------------------
__global__ void imp_dot(const unsigned short* __restrict__ u2,
                        const float* __restrict__ w3,
                        const float* __restrict__ b3,
                        float* __restrict__ impv) {
  const long r = (long)blockIdx.x * blockDim.x + threadIdx.x;
  if (r >= (long)NPTS) return;
  float acc = b3[0];
#pragma unroll 8
  for (int k = 0; k < IMPH; ++k)
    acc += bf16_to_f32(u2[r * IMPH + k]) * w3[k];
  impv[r] = acc;
}

// ---------------- Gumbel perturbation + iterative top-128 ------------------
__global__ __launch_bounds__(256) void topk_sel_kernel(
    const float* __restrict__ impv, const float* __restrict__ coords,
    const float* __restrict__ temp_in, int* __restrict__ sel,
    float* __restrict__ cent) {
  __shared__ float pert[PPTS];
  __shared__ float rmax[256];
  __shared__ int   ridx[256];
  const int b = blockIdx.x, t = threadIdx.x;
  const float temp = log1pf(expf(temp_in[0])) + 0.1f;
  for (int i = t; i < PPTS; i += 256) {
    unsigned long long x =
        (((unsigned long long)b << 32) | (unsigned)i) ^ 0x9E3779B97F4A7C15ull;
    x ^= x >> 30; x *= 0xBF58476D1CE4E5B9ull;
    x ^= x >> 27; x *= 0x94D049BB133111EBull;
    x ^= x >> 31;
    const float u01 = (float)(x >> 40) * (1.0f / 16777216.0f);
    const float g = -logf(-logf(u01 + 1e-10f) + 1e-10f);
    pert[i] = (impv[(long)b * PPTS + i] + g) / temp;
  }
  __syncthreads();
  for (int s = 0; s < MAXTOK; ++s) {
    float bm = -3.4e38f; int bi = PPTS;
    for (int i = t; i < PPTS; i += 256) {
      const float v = pert[i];
      if (v > bm || (v == bm && i < bi)) { bm = v; bi = i; }
    }
    rmax[t] = bm; ridx[t] = bi;
    __syncthreads();
    for (int off = 128; off > 0; off >>= 1) {
      if (t < off) {
        const float v = rmax[t + off]; const int j = ridx[t + off];
        if (v > rmax[t] || (v == rmax[t] && j < ridx[t])) { rmax[t] = v; ridx[t] = j; }
      }
      __syncthreads();
    }
    if (t == 0) {
      const int bidx = ridx[0];
      sel[b * MAXTOK + s] = bidx;
      pert[bidx] = -3.4e38f;
      for (int j = 0; j < 4; ++j)
        cent[((long)b * MAXTOK + s) * 4 + j] =
            coords[((long)b * PPTS + bidx) * 5 + 1 + j];
    }
    __syncthreads();
  }
}

// --------------- 4-D KNN (K=16) + max-pool of 768-d features ---------------
__global__ __launch_bounds__(256) void knn_pool_kernel(
    const float* __restrict__ cent, const float* __restrict__ coords,
    const unsigned short* __restrict__ pf, unsigned short* __restrict__ pooled) {
  __shared__ float d2[PPTS];
  __shared__ float rmin[256];
  __shared__ int   ridx[256];
  __shared__ int   knn[KNN_K];
  const int bc = blockIdx.x;
  const int b  = bc >> 7;   // MAXTOK = 128
  const int t  = threadIdx.x;
  const float cx = cent[(long)bc * 4 + 0], cy = cent[(long)bc * 4 + 1];
  const float cz = cent[(long)bc * 4 + 2], cw = cent[(long)bc * 4 + 3];
  for (int i = t; i < PPTS; i += 256) {
    const float* p = coords + ((long)b * PPTS + i) * 5 + 1;
    const float dx = cx - p[0], dy = cy - p[1], dz = cz - p[2], dw = cw - p[3];
    d2[i] = dx * dx + dy * dy + dz * dz + dw * dw;
  }
  __syncthreads();
  for (int s = 0; s < KNN_K; ++s) {
    float bm = 3.4e38f; int bi = PPTS;
    for (int i = t; i < PPTS; i += 256) {
      const float v = d2[i];
      if (v < bm || (v == bm && i < bi)) { bm = v; bi = i; }
    }
    rmin[t] = bm; ridx[t] = bi;
    __syncthreads();
    for (int off = 128; off > 0; off >>= 1) {
      if (t < off) {
        const float v = rmin[t + off]; const int j = ridx[t + off];
        if (v < rmin[t] || (v == rmin[t] && j < ridx[t])) { rmin[t] = v; ridx[t] = j; }
      }
      __syncthreads();
    }
    if (t == 0) { knn[s] = ridx[0]; d2[ridx[0]] = 3.4e38f; }
    __syncthreads();
  }
  for (int c = t; c < TOKD; c += 256) {
    float m = -3.4e38f;
#pragma unroll
    for (int j = 0; j < KNN_K; ++j)
      m = fmaxf(m, bf16_to_f32(pf[((long)b * PPTS + knn[j]) * TOKD + c]));
    pooled[(long)bc * TOKD + c] = f32_to_bf16(m);
  }
}

// -------- stable argsort by centroid time + scatter into d_out -------------
__global__ __launch_bounds__(256) void order_scatter_kernel(
    const float* __restrict__ tokf, const float* __restrict__ cent,
    float* __restrict__ out) {
  __shared__ float tv[MAXTOK];
  __shared__ int   rnk[MAXTOK];
  const int b = blockIdx.x, t = threadIdx.x;
  if (t < MAXTOK) tv[t] = cent[((long)b * MAXTOK + t) * 4 + 3];
  __syncthreads();
  if (t < MAXTOK) {
    const float ti = tv[t];
    int r = 0;
    for (int j = 0; j < MAXTOK; ++j) {
      const float tj = tv[j];
      r += (tj < ti) || (tj == ti && j < t);
    }
    rnk[t] = r;
  }
  __syncthreads();
  float* otok  = out;                                           // B*128*768
  float* ocen  = out + (long)BATCH * MAXTOK * TOKD;             // B*128*4
  float* omask = ocen + (long)BATCH * MAXTOK * 4;               // B*128
  for (int i = 0; i < MAXTOK; ++i) {
    const int dr = rnk[i];
    for (int c = t; c < TOKD; c += 256)
      otok[((long)b * MAXTOK + dr) * TOKD + c] =
          tokf[((long)b * MAXTOK + i) * TOKD + c];
  }
  if (t < MAXTOK) {
    const int dr = rnk[t];
    for (int j = 0; j < 4; ++j)
      ocen[((long)b * MAXTOK + dr) * 4 + j] = cent[((long)b * MAXTOK + t) * 4 + j];
    omask[(long)b * MAXTOK + t] = 1.0f;
  }
}

// ---------------------------------------------------------------------------
extern "C" void kernel_launch(void* const* d_in, const int* in_sizes, int n_in,
                              void* d_out, int out_size, void* d_ws, size_t ws_size,
                              hipStream_t stream) {
  (void)in_sizes; (void)n_in; (void)out_size; (void)ws_size;
  // Flattened input order (jax tree: sorted dict keys, lists in order):
  const float* coords  = (const float*)d_in[0];   // N x 5
  const float* feats   = (const float*)d_in[1];   // N x 32
  const float* imp_b1  = (const float*)d_in[2];
  const float* imp_b2  = (const float*)d_in[3];
  const float* imp_b3  = (const float*)d_in[4];
  const float* imp_lnb = (const float*)d_in[5];
  const float* imp_lng = (const float*)d_in[6];
  const float* imp_w1  = (const float*)d_in[7];   // 772 x 256
  const float* imp_w2  = (const float*)d_in[8];   // 256 x 256
  const float* imp_w3  = (const float*)d_in[9];   // 256 x 1
  const float* mlp_w0  = (const float*)d_in[10];  // 32 x 256
  const float* mlp_b0  = (const float*)d_in[11];
  const float* mlp_w1  = (const float*)d_in[12];  // 256 x 512
  const float* mlp_b1  = (const float*)d_in[13];
  const float* mlp_w2  = (const float*)d_in[14];  // 512 x 768
  const float* mlp_b2  = (const float*)d_in[15];
  const float* mlp_w3  = (const float*)d_in[16];  // 768 x 768
  const float* mlp_b3  = (const float*)d_in[17];
  const float* nbr_b1  = (const float*)d_in[18];
  const float* nbr_b2  = (const float*)d_in[19];
  const float* nbr_w1  = (const float*)d_in[20];  // 768 x 768
  const float* nbr_w2  = (const float*)d_in[21];  // 768 x 768
  const float* temper  = (const float*)d_in[22];

  // Workspace carve-up (256B aligned).
  size_t off = 0;
  char* base = (char*)d_ws;
  auto carve = [&](size_t bytes) -> char* {
    char* p = base + off;
    off += (bytes + 255) & ~(size_t)255;
    return p;
  };
  unsigned short* featb = (unsigned short*)carve((size_t)NPTS * FEAT * 2);
  unsigned short* bw0   = (unsigned short*)carve((size_t)32 * 256 * 2);
  unsigned short* bw1   = (unsigned short*)carve((size_t)256 * 512 * 2);
  unsigned short* bw2   = (unsigned short*)carve((size_t)512 * 768 * 2);
  unsigned short* bw3   = (unsigned short*)carve((size_t)768 * 768 * 2);
  unsigned short* biw1  = (unsigned short*)carve((size_t)772 * 256 * 2);
  unsigned short* biw2  = (unsigned short*)carve((size_t)256 * 256 * 2);
  unsigned short* bnw1  = (unsigned short*)carve((size_t)768 * 768 * 2);
  unsigned short* bnw2  = (unsigned short*)carve((size_t)768 * 768 * 2);
  // Region R1 (N*768 bf16): h1 (N*256) | h2 (N*512); later reused as pf.
  char* R1 = carve((size_t)NPTS * TOKD * 2);
  unsigned short* h1 = (unsigned short*)R1;
  unsigned short* h2 = (unsigned short*)(R1 + (size_t)NPTS * 256 * 2);
  unsigned short* pf = (unsigned short*)R1;            // reuse after h1,h2 dead
  // Region R2 (N*768 bf16 bytes): h3; later u1(f32, N*256) + u1b(bf16) + u2.
  char* R2 = carve((size_t)NPTS * TOKD * 2);
  unsigned short* h3  = (unsigned short*)R2;
  float*          u1  = (float*)R2;                              // N*256 f32
  unsigned short* u1b = (unsigned short*)(R2 + (size_t)NPTS * 256 * 4);
  unsigned short* u2  = (unsigned short*)R2;                     // reuse of u1
  float* impv   = (float*)carve((size_t)NPTS * 4);
  int*   sel    = (int*)carve((size_t)BATCH * MAXTOK * 4);
  float* centw  = (float*)carve((size_t)BATCH * MAXTOK * 4 * 4);
  unsigned short* pooled = (unsigned short*)carve((size_t)BATCH * MAXTOK * TOKD * 2);
  unsigned short* t1     = (unsigned short*)carve((size_t)BATCH * MAXTOK * TOKD * 2);
  float* tokf   = (float*)carve((size_t)BATCH * MAXTOK * TOKD * 4);

  // 0) one-time f32 -> bf16 conversions.
  cvt_f32_bf16<<<2048, 256, 0, stream>>>(feats,  featb, (long)NPTS * FEAT);
  cvt_f32_bf16<<<64,   256, 0, stream>>>(mlp_w0, bw0,  32L * 256);
  cvt_f32_bf16<<<512,  256, 0, stream>>>(mlp_w1, bw1,  256L * 512);
  cvt_f32_bf16<<<1024, 256, 0, stream>>>(mlp_w2, bw2,  512L * 768);
  cvt_f32_bf16<<<1024, 256, 0, stream>>>(mlp_w3, bw3,  768L * 768);
  cvt_f32_bf16<<<512,  256, 0, stream>>>(imp_w1, biw1, 772L * 256);
  cvt_f32_bf16<<<256,  256, 0, stream>>>(imp_w2, biw2, 256L * 256);
  cvt_f32_bf16<<<1024, 256, 0, stream>>>(nbr_w1, bnw1, 768L * 768);
  cvt_f32_bf16<<<1024, 256, 0, stream>>>(nbr_w2, bnw2, 768L * 768);

  const dim3 blk(256);
  // 1) point MLP: 32 -> 256 -> 512 -> 768 (relu) -> 768 (linear) => pf
  gemm_bf16_wmma<<<dim3(256 / TN, NPTS / TM), blk, 0, stream>>>(
      featb, bw0, mlp_b0, nullptr, nullptr, h1, nullptr, 32, 256, 1);
  gemm_bf16_wmma<<<dim3(512 / TN, NPTS / TM), blk, 0, stream>>>(
      h1, bw1, mlp_b1, nullptr, nullptr, h2, nullptr, 256, 512, 1);
  gemm_bf16_wmma<<<dim3(768 / TN, NPTS / TM), blk, 0, stream>>>(
      h2, bw2, mlp_b2, nullptr, nullptr, h3, nullptr, 512, 768, 1);
  gemm_bf16_wmma<<<dim3(768 / TN, NPTS / TM), blk, 0, stream>>>(
      h3, bw3, mlp_b3, nullptr, nullptr, pf, nullptr, 768, 768, 0);

  // 2) importance net: concat([pf, xyzt]) @ w1 (+relu) -> LN -> @w2 (+relu) -> dot w3
  gemm_bf16_wmma<<<dim3(IMPH / TN, NPTS / TM), blk, 0, stream>>>(
      pf, biw1, imp_b1, coords, imp_w1 + 768L * 256, nullptr, u1, 768, IMPH, 1);
  ln256<<<NPTS, 256, 0, stream>>>(u1, imp_lng, imp_lnb, u1b);
  gemm_bf16_wmma<<<dim3(IMPH / TN, NPTS / TM), blk, 0, stream>>>(
      u1b, biw2, imp_b2, nullptr, nullptr, u2, nullptr, IMPH, IMPH, 1);
  imp_dot<<<NPTS / 256, 256, 0, stream>>>(u2, imp_w3, imp_b3, impv);

  // 3) Gumbel top-128 per event, 4-D KNN + max-pool.
  topk_sel_kernel<<<BATCH, 256, 0, stream>>>(impv, coords, temper, sel, centw);
  knn_pool_kernel<<<BATCH * MAXTOK, 256, 0, stream>>>(centw, coords, pf, pooled);

  // 4) token MLP: relu(pooled @ nw1 + b1) @ nw2 + b2
  gemm_bf16_wmma<<<dim3(768 / TN, (BATCH * MAXTOK) / TM), blk, 0, stream>>>(
      pooled, bnw1, nbr_b1, nullptr, nullptr, t1, nullptr, 768, 768, 1);
  gemm_bf16_wmma<<<dim3(768 / TN, (BATCH * MAXTOK) / TM), blk, 0, stream>>>(
      t1, bnw2, nbr_b2, nullptr, nullptr, nullptr, tokf, 768, 768, 0);

  // 5) sort by centroid time, scatter tokens/centroids/masks into d_out.
  order_scatter_kernel<<<BATCH, 256, 0, stream>>>(tokf, centw, (float*)d_out);
}
